// MaskCrossAttention_49417893708129
// MI455X (gfx1250) — matrix operand order repaired and checked by reading, not compiled, not run on hardware
//
#include <hip/hip_runtime.h>
#include <hip/hip_bf16.h>

// ---------------------------------------------------------------------------
// Types
// ---------------------------------------------------------------------------
typedef __bf16 bf16_t;
typedef __attribute__((ext_vector_type(16))) __bf16 v16bf;
typedef __attribute__((ext_vector_type(8)))  float  v8f;
typedef __attribute__((ext_vector_type(4)))  unsigned int u32x4;
typedef __attribute__((ext_vector_type(8)))  int i32x8;
typedef __attribute__((ext_vector_type(4)))  int i32x4;

union Frag16 {
  v16bf v;
  u32x4 q[2];
  unsigned short u[16];
};

// ---------------------------------------------------------------------------
// Problem constants
// ---------------------------------------------------------------------------
#define BATCH 4
#define LQ    1024
#define LKV   1024
#define EMB   768
#define NHEAD 12
#define HD    64
#define QK_SCALE 0.125f   // 1/sqrt(64)
#define KVSTEP 64
#define NSTEP (LKV / KVSTEP)   // 16

// ---------------------------------------------------------------------------
// Helpers
// ---------------------------------------------------------------------------
__device__ __forceinline__ unsigned short f2bf_raw(float x) {
  unsigned int u = __float_as_uint(x);
  u += 0x7FFFu + ((u >> 16) & 1u);   // round-to-nearest-even
  return (unsigned short)(u >> 16);
}
__device__ __forceinline__ bf16_t f2bf(float x) {
  unsigned short h = f2bf_raw(x);
  return __builtin_bit_cast(__bf16, h);
}

__device__ __forceinline__ v8f zero8() {
  v8f z;
#pragma unroll
  for (int i = 0; i < 8; ++i) z[i] = 0.f;
  return z;
}

// A-matrix 16x32 bf16 fragment (ISA 7.12.2): lane m = lane%16, half = lane/16,
// per-lane K values: [half*8, half*8+8) then [16+half*8, 16+half*8+8).
__device__ __forceinline__ void load_fragA(Frag16& f, const bf16_t* row, int half) {
  f.q[0] = *reinterpret_cast<const u32x4*>(row + half * 8);
  f.q[1] = *reinterpret_cast<const u32x4*>(row + 16 + half * 8);
}

// B-matrix 32x16 bf16 fragment (ISA 7.12.5 B layout): lane n = lane%16,
// per-lane K values: [half*8, half*8+8) then [16+half*8, 16+half*8+8).
// `row` points at column n of B stored K-contiguous (i.e. B^T row n).
__device__ __forceinline__ void load_fragB(Frag16& f, const bf16_t* row, int half) {
  f.q[0] = *reinterpret_cast<const u32x4*>(row + half * 8);
  f.q[1] = *reinterpret_cast<const u32x4*>(row + 16 + half * 8);
}

__device__ __forceinline__ v8f wmma_bf16(const Frag16& a, const Frag16& b, v8f c) {
  return __builtin_amdgcn_wmma_f32_16x16x32_bf16(
      /*neg_a=*/false, a.v, /*neg_b=*/false, b.v,
      /*c_mod=*/(short)0, c, /*reuse_a=*/false, /*reuse_b=*/false);
}

// ---------------------------------------------------------------------------
// Tensor Data Mover: async 2-D tile load Global -> LDS (D# per ISA 08 §8).
//   data_size_code: 0=1B,1=2B,2=4B,3=8B. Strides/dims in elements.
//   Optional LDS row padding: after (2^(pad_interval+1)) DWORDs stored,
//   skip (pad_amount+1) DWORDs in LDS.
// ---------------------------------------------------------------------------
__device__ __forceinline__ void tdm_load_2d(
    unsigned lds_off, const void* gaddr,
    unsigned tile_x, unsigned tile_y,
    unsigned tensor_x, unsigned tensor_y,
    unsigned stride_x, unsigned data_size_code,
    unsigned pad_enable, unsigned pad_interval, unsigned pad_amount)
{
  const unsigned long long ga = (unsigned long long)(uintptr_t)gaddr;
  u32x4 g0;
  g0[0] = 1u;                                       // count=1 (valid, user mode)
  g0[1] = lds_off;                                  // lds_addr
  g0[2] = (unsigned)ga;                             // global_addr[31:0]
  g0[3] = (unsigned)((ga >> 32) & 0x01FFFFFFu)      // global_addr[56:32]
          | (2u << 30);                             // type=2 ("image")
  i32x8 g1;
  g1[0] = (int)((data_size_code << 16) | (pad_enable << 20) |
                (pad_interval << 22) | (pad_amount << 25));
  g1[1] = (int)(tensor_x << 16);                    // tensor_dim0[15:0] @ bit48
  g1[2] = (int)((tensor_x >> 16) | (tensor_y << 16));
  g1[3] = (int)((tensor_y >> 16) | (tile_x << 16)); // tile_dim0 @ bit112
  g1[4] = (int)(tile_y & 0xFFFFu);                  // tile_dim1; tile_dim2=0
  g1[5] = (int)stride_x;                            // tensor_dim0_stride[31:0]
  g1[6] = 0;                                        // stride0[47:32] | stride1 lo
  g1[7] = 0;
  const i32x4 z4 = {0, 0, 0, 0};
#if defined(__clang_major__) && __clang_major__ >= 23
  const i32x8 z8 = {0, 0, 0, 0, 0, 0, 0, 0};
  __builtin_amdgcn_tensor_load_to_lds(g0, g1, z4, z4, z8, 0);
#else
  __builtin_amdgcn_tensor_load_to_lds(g0, g1, z4, z4, 0);
#endif
}

__device__ __forceinline__ unsigned lds_offset_of(const void* p) {
  return (unsigned)(uintptr_t)p;   // flat LDS aperture: low 32 bits = LDS offset
}

// ---------------------------------------------------------------------------
// Generic GEMM: C[M,N] = (A[M,K] @ W[K,N] + bias[N]) * scale
//   A: fp32 or bf16 row-major.  W: fp32 row-major.  C: fp32 or bf16.
//   Block: 256 threads (8 waves). Block tile 128x64; wave tile 32x32.
// ---------------------------------------------------------------------------
template <bool A_BF16, bool OUT_BF16>
__global__ __launch_bounds__(256) void gemm_wmma_kernel(
    const void* __restrict__ Ap, const float* __restrict__ W,
    const float* __restrict__ bias, void* __restrict__ Cp,
    int M, int N, int K, float scale)
{
  __shared__ bf16_t As[128][40];  // rows m, cols k   (pitch 80B, 16B aligned)
  __shared__ bf16_t Bs[64][40];   // rows n, cols k   (W transposed)

  const int t    = threadIdx.x;
  const int wave = t >> 5;
  const int lane = t & 31;
  const int half = lane >> 4;
  const int ln   = lane & 15;

  const int m0 = blockIdx.y * 128;
  const int n0 = blockIdx.x * 64;
  const int wm = (wave >> 1) * 32;  // wave row offset in block tile
  const int wn = (wave & 1) * 32;   // wave col offset in block tile

  v8f acc[2][2];
#pragma unroll
  for (int i = 0; i < 2; ++i)
#pragma unroll
    for (int j = 0; j < 2; ++j) acc[i][j] = zero8();

  // cooperative-load assignments
  const int arow = t >> 1;          // 0..127
  const int acol = (t & 1) * 16;    // 0 or 16
  const int bk   = t >> 3;          // 0..31
  const int bn   = (t & 7) * 8;     // 0..56

  for (int k0 = 0; k0 < K; k0 += 32) {
    // --- stage A tile (convert to bf16 if needed) ---
    if constexpr (A_BF16) {
      const bf16_t* src = (const bf16_t*)Ap + (size_t)(m0 + arow) * K + k0 + acol;
      u32x4* dst = reinterpret_cast<u32x4*>(&As[arow][acol]);
      const u32x4* s4 = reinterpret_cast<const u32x4*>(src);
      dst[0] = s4[0];
      dst[1] = s4[1];
    } else {
      const float* src = (const float*)Ap + (size_t)(m0 + arow) * K + k0 + acol;
#pragma unroll
      for (int j = 0; j < 16; ++j) As[arow][acol + j] = f2bf(src[j]);
    }
    // --- stage B tile transposed: Bs[n][k] = W[k0+k][n0+n] ---
    {
      const float* src = W + (size_t)(k0 + bk) * N + n0 + bn;
#pragma unroll
      for (int j = 0; j < 8; ++j) Bs[bn + j][bk] = f2bf(src[j]);
    }
    __syncthreads();

    Frag16 af[2], bfm[2];
    load_fragA(af[0], &As[wm + ln][0], half);
    load_fragA(af[1], &As[wm + 16 + ln][0], half);
    load_fragB(bfm[0], &Bs[wn + ln][0], half);
    load_fragB(bfm[1], &Bs[wn + 16 + ln][0], half);

#pragma unroll
    for (int mt = 0; mt < 2; ++mt)
#pragma unroll
      for (int nt = 0; nt < 2; ++nt)
        acc[mt][nt] = wmma_bf16(af[mt], bfm[nt], acc[mt][nt]);

    __syncthreads();
  }

  // epilogue: C layout lane (n = ln), vgpr r -> row m = r + 8*half
#pragma unroll
  for (int mt = 0; mt < 2; ++mt)
#pragma unroll
    for (int nt = 0; nt < 2; ++nt) {
      const int col = n0 + wn + nt * 16 + ln;
      const float bv = bias[col];
#pragma unroll
      for (int r = 0; r < 8; ++r) {
        const int row = m0 + wm + mt * 16 + r + 8 * half;
        const float v = (acc[mt][nt][r] + bv) * scale;
        if constexpr (OUT_BF16)
          ((bf16_t*)Cp)[(size_t)row * N + col] = f2bf(v);
        else
          ((float*)Cp)[(size_t)row * N + col] = v;
      }
    }
}

// ---------------------------------------------------------------------------
// Flash attention over one (b, h, 128-query tile), TDM double-buffered.
//   Qbf:  (B, LQ, EMB)  bf16, already scaled by QK_SCALE
//   KVbf: (B, LKV, 2*EMB) bf16 : [0:768 K all heads | 768:1536 V all heads]
//   mask: (B, LQ) fp32 {0,1};  attn_pos: (NH, LQ, LKV) fp32
//   Obf:  (B, LQ, EMB) bf16
// ---------------------------------------------------------------------------
__global__ __launch_bounds__(256) void attn_flash_kernel(
    const bf16_t* __restrict__ Qbf, const bf16_t* __restrict__ KVbf,
    const float* __restrict__ mask, const float* __restrict__ attn_pos,
    bf16_t* __restrict__ Obf)
{
  __shared__ bf16_t Ks[2][64][72];     // [buf][kv][d]  (TDM, padded rows)
  __shared__ bf16_t Vs[64][72];        // [d][kv]       (manual transpose)
  __shared__ bf16_t Ps[8][16][72];     // per-wave P scratch [row][kv]
  __shared__ float  PosS[2][128][64];  // [buf][q_local][kv_local] (TDM)

  const int t    = threadIdx.x;
  const int wave = t >> 5;
  const int lane = t & 31;
  const int half = lane >> 4;
  const int ln   = lane & 15;

  const int q0 = blockIdx.x * 128;
  const int h  = blockIdx.y;
  const int b  = blockIdx.z;

  // --- per-row softmax state (rows owned by this lane: q = qbase + r + 8*half)
  const int qbase = q0 + wave * 16;
  float m_run[8], l_run[8], rowbias[8];
#pragma unroll
  for (int r = 0; r < 8; ++r) {
    const int q = qbase + r + 8 * half;
    const float mv = mask[(size_t)b * LQ + q];
    rowbias[r] = (1.f - mv) * -100.f;
    // dummy column: logit 100 if masked else -100; contributes exp(dummy-m)=1
    m_run[r] = (mv == 0.f) ? 100.f : -100.f;
    l_run[r] = 1.f;
  }

  // --- Q fragments (held for the whole kernel): row m = ln of this wave's strip
  Frag16 aQ[2];
  {
    const bf16_t* qrow = Qbf + ((size_t)(b * LQ + qbase + ln)) * EMB + h * HD;
    load_fragA(aQ[0], qrow, half);       // d = 0..31
    load_fragA(aQ[1], qrow + 32, half);  // d = 32..63
  }

  v8f acc[4];
#pragma unroll
  for (int d = 0; d < 4; ++d) acc[d] = zero8();

  // TDM issue helper (uniform addresses; wave 0 only)
  const bf16_t* Kg0  = KVbf + ((size_t)(b * LKV)) * (2 * EMB) + h * HD;
  const float*  Pg0  = attn_pos + ((size_t)(h * LQ + q0)) * LKV;
  auto issue_step = [&](int step, int buf) {
    // K tile: 64(d) x 64(kv) bf16, row stride 1536 elems; pad 128B rows -> 144B
    tdm_load_2d(lds_offset_of(&Ks[buf][0][0]),
                Kg0 + (size_t)step * KVSTEP * (2 * EMB),
                /*tile_x=*/HD, /*tile_y=*/64, /*tensor_x=*/HD, /*tensor_y=*/64,
                /*stride_x=*/2 * EMB, /*dsz=*/1,
                /*pad_en=*/1, /*pad_interval=*/4, /*pad_amount=*/3);
    // attn_pos tile: 64(kv) x 128(q) fp32, row stride 1024 elems, no padding
    tdm_load_2d(lds_offset_of(&PosS[buf][0][0]),
                Pg0 + step * KVSTEP,
                /*tile_x=*/KVSTEP, /*tile_y=*/128, /*tensor_x=*/KVSTEP,
                /*tensor_y=*/128, /*stride_x=*/LKV, /*dsz=*/2,
                /*pad_en=*/0, /*pad_interval=*/0, /*pad_amount=*/0);
  };

  if (wave == 0) issue_step(0, 0);   // prologue: prime buffer 0

  // cooperative V staging assignment: 4 threads per kv row, 16 d each
  const int krow = t >> 2;         // 0..63
  const int kcol = (t & 3) * 16;   // 0,16,32,48

  for (int it = 0; it < NSTEP; ++it) {
    const int kv0 = it * KVSTEP;
    const int buf = it & 1;

    __syncthreads();  // all readers of both buffers (prev iters) are done

    if (wave == 0) {
      if (it + 1 < NSTEP) {
        issue_step(it + 1, buf ^ 1);               // async: overlaps compute
        __builtin_amdgcn_s_wait_tensorcnt(2);      // current buf's pair done
      } else {
        __builtin_amdgcn_s_wait_tensorcnt(0);
      }
    }

    // --- stage V tile transposed: Vs[d][kv] (manual; TDM can't transpose) ---
    {
      const bf16_t* src =
          KVbf + ((size_t)(b * LKV + kv0 + krow)) * (2 * EMB) + EMB + h * HD + kcol;
      Frag16 tmp;
      tmp.q[0] = *reinterpret_cast<const u32x4*>(src);
      tmp.q[1] = *reinterpret_cast<const u32x4*>(src + 8);
#pragma unroll
      for (int j = 0; j < 16; ++j) Vs[kcol + j][krow] = (bf16_t)tmp.v[j];
    }
    if (it + 1 < NSTEP) {
      const bf16_t* nv = KVbf +
          ((size_t)(b * LKV + kv0 + KVSTEP + krow)) * (2 * EMB) + EMB + h * HD + kcol;
      __builtin_prefetch(nv, 0, 1);   // gfx1250 global_prefetch path
    }
    __syncthreads();

    // --- S = Q*K^T : 4 kv n-tiles x 2 K-chunks of 32 ---
    v8f s[4];
#pragma unroll
    for (int nt = 0; nt < 4; ++nt) {
      Frag16 bk;
      load_fragB(bk, &Ks[buf][nt * 16 + ln][0], half);
      v8f p = wmma_bf16(aQ[0], bk, zero8());
      load_fragB(bk, &Ks[buf][nt * 16 + ln][32], half);
      s[nt] = wmma_bf16(aQ[1], bk, p);
    }

    // --- add attn_pos (from LDS) and mask bias ---
#pragma unroll
    for (int nt = 0; nt < 4; ++nt) {
#pragma unroll
      for (int r = 0; r < 8; ++r) {
        const float ap = PosS[buf][wave * 16 + r + 8 * half][nt * 16 + ln];
        s[nt][r] = s[nt][r] + ap + rowbias[r];
      }
    }

    // --- online softmax (row = 16 lanes of a half-wave) ---
#pragma unroll
    for (int r = 0; r < 8; ++r) {
      float mx = fmaxf(fmaxf(s[0][r], s[1][r]), fmaxf(s[2][r], s[3][r]));
      mx = fmaxf(mx, __shfl_xor(mx, 1));
      mx = fmaxf(mx, __shfl_xor(mx, 2));
      mx = fmaxf(mx, __shfl_xor(mx, 4));
      mx = fmaxf(mx, __shfl_xor(mx, 8));
      const float mnew = fmaxf(m_run[r], mx);
      const float alpha = __expf(m_run[r] - mnew);
      float rs = 0.f;
#pragma unroll
      for (int nt = 0; nt < 4; ++nt) {
        const float p = __expf(s[nt][r] - mnew);
        s[nt][r] = p;
        rs += p;
      }
      rs += __shfl_xor(rs, 1);
      rs += __shfl_xor(rs, 2);
      rs += __shfl_xor(rs, 4);
      rs += __shfl_xor(rs, 8);
      l_run[r] = l_run[r] * alpha + rs;
      m_run[r] = mnew;
#pragma unroll
      for (int d = 0; d < 4; ++d) acc[d][r] *= alpha;
    }

    // --- P -> LDS (reshape C-layout -> A-layout) ---
#pragma unroll
    for (int nt = 0; nt < 4; ++nt)
#pragma unroll
      for (int r = 0; r < 8; ++r)
        Ps[wave][r + 8 * half][nt * 16 + ln] = f2bf(s[nt][r]);

    __syncthreads();

    Frag16 aP0, aP1;
    load_fragA(aP0, &Ps[wave][ln][0], half);   // kv = 0..31
    load_fragA(aP1, &Ps[wave][ln][32], half);  // kv = 32..63

    // --- O += P * V : 4 d n-tiles x 2 kv K-chunks ---
#pragma unroll
    for (int dt = 0; dt < 4; ++dt) {
      Frag16 bv;
      load_fragB(bv, &Vs[dt * 16 + ln][0], half);
      acc[dt] = wmma_bf16(aP0, bv, acc[dt]);
      load_fragB(bv, &Vs[dt * 16 + ln][32], half);
      acc[dt] = wmma_bf16(aP1, bv, acc[dt]);
    }
  }

  // --- normalize and write O (bf16, (B,LQ,EMB) layout) ---
  float inv[8];
#pragma unroll
  for (int r = 0; r < 8; ++r) inv[r] = 1.f / l_run[r];
#pragma unroll
  for (int dt = 0; dt < 4; ++dt)
#pragma unroll
    for (int r = 0; r < 8; ++r) {
      const int q = qbase + r + 8 * half;
      Obf[((size_t)(b * LQ + q)) * EMB + h * HD + dt * 16 + ln] =
          f2bf(acc[dt][r] * inv[r]);
    }
}

// ---------------------------------------------------------------------------
// Launch
// ---------------------------------------------------------------------------
extern "C" void kernel_launch(void* const* d_in, const int* in_sizes, int n_in,
                              void* d_out, int out_size, void* d_ws, size_t ws_size,
                              hipStream_t stream) {
  const float* Xq       = (const float*)d_in[0];  // (B,LQ,CQ)
  const float* Xkv      = (const float*)d_in[1];  // (B,LKV,CKV)
  const float* mask     = (const float*)d_in[2];  // (B,LQ,1)
  const float* Wq       = (const float*)d_in[3];  // (CQ,EMB)
  const float* bq       = (const float*)d_in[4];  // (EMB)
  const float* Wkv      = (const float*)d_in[5];  // (CKV,2*EMB)
  const float* bkv      = (const float*)d_in[6];  // (2*EMB)
  const float* Wp       = (const float*)d_in[7];  // (EMB,EMB)
  const float* bp       = (const float*)d_in[8];  // (EMB)
  const float* attn_pos = (const float*)d_in[9];  // (1,NH,LQ,LKV)

  // workspace layout (bf16 intermediates)
  const size_t Q_BYTES  = (size_t)BATCH * LQ * EMB * 2;        //  6 MB
  const size_t KV_BYTES = (size_t)BATCH * LKV * 2 * EMB * 2;   // 12 MB
  bf16_t* Qbf  = (bf16_t*)d_ws;
  bf16_t* KVbf = (bf16_t*)((char*)d_ws + Q_BYTES);
  bf16_t* Obf  = (bf16_t*)((char*)d_ws + Q_BYTES + KV_BYTES);

  const dim3 blk(256);
  const int M = BATCH * LQ;  // 4096

  // Q = (Xq@Wq + bq) * SCALE  -> bf16
  gemm_wmma_kernel<false, true><<<dim3(EMB / 64, M / 128), blk, 0, stream>>>(
      Xq, Wq, bq, Qbf, M, EMB, EMB, QK_SCALE);
  // KV = Xkv@Wkv + bkv -> bf16
  gemm_wmma_kernel<false, true><<<dim3((2 * EMB) / 64, M / 128), blk, 0, stream>>>(
      Xkv, Wkv, bkv, KVbf, M, 2 * EMB, EMB, 1.0f);
  // flash attention (TDM double-buffered K + attn_pos tiles)
  attn_flash_kernel<<<dim3(LQ / 128, NHEAD, BATCH), blk, 0, stream>>>(
      Qbf, KVbf, mask, attn_pos, Obf);
  // out = O@Wp + bp -> fp32
  gemm_wmma_kernel<true, false><<<dim3(EMB / 64, M / 128), blk, 0, stream>>>(
      Obf, Wp, bp, (float*)d_out, M, EMB, EMB, 1.0f);
}
